// LinSinkhornRegModel_21371757265162
// MI455X (gfx1250) — matrix operand order repaired and checked by reading, not compiled, not run on hardware
//
#include <hip/hip_runtime.h>
#include <hip/hip_bf16.h>

// ---------------------------------------------------------------------------
// Sinkhorn divergence (geomloss-style, debiased) for MI455X / gfx1250.
// bf16 feature matrices stay L2-resident (X: 12.6MB, Y: 3.1MB << 192MB L2);
// cost tiles are recomputed with v_wmma_f32_16x16x32_bf16 inside a fused
// online-logsumexp softmin. Two accumulator chains + K-loop unrolling give
// WMMA->WMMA spacing; per-lane streaming stats, merged once at the end.
// ---------------------------------------------------------------------------

#define DIMD 768
#define NN   8192
#define MM   2048
#define EPS_F    0.0025f     // blur^p = 0.05^2
#define INV_EPS  400.0f      // 1/eps

typedef __attribute__((ext_vector_type(16))) __bf16 v16bf;
typedef __attribute__((ext_vector_type(8)))  __bf16 v8bf;
typedef __attribute__((ext_vector_type(8)))  float  v8f;
typedef __attribute__((ext_vector_type(4)))  float  v4f;

static __device__ __forceinline__ v16bf cat8(v8bf lo, v8bf hi) {
  return __builtin_shufflevector(lo, hi, 0, 1, 2, 3, 4, 5, 6, 7,
                                 8, 9, 10, 11, 12, 13, 14, 15);
}

// ---------------------------------------------------------------------------
// Transform GEMM: Out[r][c] = sum_k A[r][k] * W[c][k]   (x = d @ W^T)
// f32 inputs converted to bf16 in-register; bf16 output (consumed by softmin).
// ---------------------------------------------------------------------------
__global__ __launch_bounds__(32) void gemm_xWt_bf16(
    const float* __restrict__ A, const float* __restrict__ W,
    __bf16* __restrict__ Out) {
  const int lane  = threadIdx.x & 31;
  const int rowBase = blockIdx.x * 16;
  const int mrow  = rowBase + (lane & 15);
  const int koffA = (lane < 16) ? 0 : 8;   // A-fragment K start (16-bit layout)
  const int ncol  = lane & 15;             // B-fragment column
  const int khB   = (lane < 16) ? 0 : 8;   // B-fragment K start (16-bit layout)
  const int rhalf = (lane < 16) ? 0 : 8;   // C/D row half

  for (int colBase = 0; colBase < DIMD; colBase += 16) {
    v8f acc = {};
    for (int k0 = 0; k0 < DIMD; k0 += 32) {
      v16bf a, b;
      // A: lane holds row mrow, K = koffA+0..7 and koffA+16..23
      const float* ap = A + (size_t)mrow * DIMD + k0 + koffA;
      const v4f* ap4 = (const v4f*)ap;
      v4f a0 = ap4[0], a1 = ap4[1], a2 = ap4[4], a3 = ap4[5];
#pragma unroll
      for (int i = 0; i < 4; ++i) {
        a[i]      = (__bf16)a0[i];
        a[4 + i]  = (__bf16)a1[i];
        a[8 + i]  = (__bf16)a2[i];
        a[12 + i] = (__bf16)a3[i];
      }
      // B: lane holds column (colBase+ncol), K = khB+0..7 and khB+16..23
      const float* bp = W + (size_t)(colBase + ncol) * DIMD + k0 + khB;
      const v4f* bp4 = (const v4f*)bp;
      v4f b0 = bp4[0], b1 = bp4[1], b2 = bp4[4], b3 = bp4[5];
#pragma unroll
      for (int i = 0; i < 4; ++i) {
        b[i]      = (__bf16)b0[i];
        b[4 + i]  = (__bf16)b1[i];
        b[8 + i]  = (__bf16)b2[i];
        b[12 + i] = (__bf16)b3[i];
      }
      acc = __builtin_amdgcn_wmma_f32_16x16x32_bf16(
          false, a, false, b, (short)0, acc, false, false);
    }
    const int orow = rowBase + rhalf;
#pragma unroll
    for (int v = 0; v < 8; ++v)
      Out[(size_t)(orow + v) * DIMD + colBase + ncol] = (__bf16)acc[v];
  }
}

// ---------------------------------------------------------------------------
// Fused softmin: out[i] = -eps * logsumexp_j( logw[j] + pot[j]/eps - C_ij/eps )
// with C_ij = 0.5*relu(xsq_i + ysq_j - 2 * x_i . y_j), dot via bf16 WMMA.
// Two 16-col tiles per outer step (shared A fragment, dual accumulators);
// per-lane streaming (m, ssum) stats; single cross-lane merge at the end.
// One wave per 16-row strip.
// ---------------------------------------------------------------------------
__global__ __launch_bounds__(32) void softmin_rows(
    const __bf16* __restrict__ X, const float* __restrict__ xsq,
    const __bf16* __restrict__ Y, const float* __restrict__ ysq, int Ccols,
    const float* __restrict__ logw, const float* __restrict__ pot,
    float* __restrict__ out) {
  const int lane  = threadIdx.x & 31;
  const int rowBase = blockIdx.x * 16;
  const int arow  = rowBase + (lane & 15);
  const int koffA = (lane < 16) ? 0 : 8;
  const int ncol  = lane & 15;
  const int khB   = (lane < 16) ? 0 : 8;
  const int rhalf = (lane < 16) ? 0 : 8;

  float xs[8], m[8], ssum[8];
  const float NEGINF = -__builtin_inff();
#pragma unroll
  for (int v = 0; v < 8; ++v) {
    xs[v] = xsq[rowBase + rhalf + v];
    m[v] = NEGINF;
    ssum[v] = 0.0f;
  }

  const __bf16* xrow = X + (size_t)arow * DIMD + koffA;

  for (int colBase = 0; colBase < Ccols; colBase += 32) {
    const int col0 = colBase + ncol;
    const int col1 = col0 + 16;
    const __bf16* ycol0 = Y + (size_t)col0 * DIMD + khB;
    const __bf16* ycol1 = Y + (size_t)col1 * DIMD + khB;
    // prefetch the next pair of column tiles (streamed operand)
    if (colBase + 32 < Ccols) {
      __builtin_prefetch(ycol0 + 32 * DIMD, 0, 1);
      __builtin_prefetch(ycol1 + 32 * DIMD, 0, 1);
    }
    v8f acc0 = {}, acc1 = {};
    for (int k0 = 0; k0 < DIMD; k0 += 32) {
      v16bf a = cat8(*(const v8bf*)(xrow + k0),
                     *(const v8bf*)(xrow + k0 + 16));
      v16bf b0 = cat8(*(const v8bf*)(ycol0 + k0), *(const v8bf*)(ycol0 + k0 + 16));
      v16bf b1 = cat8(*(const v8bf*)(ycol1 + k0), *(const v8bf*)(ycol1 + k0 + 16));
      acc0 = __builtin_amdgcn_wmma_f32_16x16x32_bf16(
          false, a, false, b0, (short)0, acc0, false, false);
      acc1 = __builtin_amdgcn_wmma_f32_16x16x32_bf16(
          false, a, false, b1, (short)0, acc1, false, false);
    }
    const float hl0 = logw[col0] + pot[col0] * INV_EPS;
    const float hl1 = logw[col1] + pot[col1] * INV_EPS;
    const float yq0 = ysq[col0];
    const float yq1 = ysq[col1];
#pragma unroll
    for (int v = 0; v < 8; ++v) {
      // per-lane online logsumexp update: 1 exp, no shuffles
      {
        float cst = 0.5f * fmaxf(xs[v] + yq0 - 2.0f * acc0[v], 0.0f);
        float s   = hl0 - cst * INV_EPS;
        float mo  = m[v];
        float mn  = fmaxf(mo, s);
        float e   = __expf(fminf(s, mo) - mn);
        ssum[v]   = (s > mo) ? (ssum[v] * e + 1.0f) : (ssum[v] + e);
        m[v]      = mn;
      }
      {
        float cst = 0.5f * fmaxf(xs[v] + yq1 - 2.0f * acc1[v], 0.0f);
        float s   = hl1 - cst * INV_EPS;
        float mo  = m[v];
        float mn  = fmaxf(mo, s);
        float e   = __expf(fminf(s, mo) - mn);
        ssum[v]   = (s > mo) ? (ssum[v] * e + 1.0f) : (ssum[v] + e);
        m[v]      = mn;
      }
    }
  }

  // single cross-lane merge of (m, ssum) pairs within each 16-lane half-group
#pragma unroll
  for (int v = 0; v < 8; ++v) {
#pragma unroll
    for (int mask = 1; mask <= 8; mask <<= 1) {
      float mo = __shfl_xor(m[v], mask);
      float so = __shfl_xor(ssum[v], mask);
      float mn = fmaxf(m[v], mo);
      ssum[v] = ssum[v] * __expf(m[v] - mn) + so * __expf(mo - mn);
      m[v] = mn;
    }
  }
  if (ncol == 0) {
#pragma unroll
    for (int v = 0; v < 8; ++v)
      out[rowBase + rhalf + v] = -EPS_F * (m[v] + __logf(ssum[v]));
  }
}

// ---------------------------------------------------------------------------
// Small elementwise helpers
// ---------------------------------------------------------------------------
__global__ void rowsq_kernel(const __bf16* __restrict__ X, int R,
                             float* __restrict__ out) {
  int r = blockIdx.x * blockDim.x + threadIdx.x;
  if (r >= R) return;
  const __bf16* row = X + (size_t)r * DIMD;
  float s = 0.0f;
  for (int k = 0; k < DIMD; k += 8) {
    v8bf t = *(const v8bf*)(row + k);
#pragma unroll
    for (int i = 0; i < 8; ++i) { float v = (float)t[i]; s += v * v; }
  }
  out[r] = s;
}

__global__ void vlog_kernel(const float* __restrict__ in, float* __restrict__ out, int n) {
  int i = blockIdx.x * blockDim.x + threadIdx.x;
  if (i < n) out[i] = __logf(in[i]);
}

__global__ void fill_kernel(float* __restrict__ p, int n, float v) {
  int i = blockIdx.x * blockDim.x + threadIdx.x;
  if (i < n) p[i] = v;
}

__global__ void damp_kernel(float* __restrict__ f, const float* __restrict__ t, int n) {
  int i = blockIdx.x * blockDim.x + threadIdx.x;
  if (i < n) f[i] = 0.5f * (f[i] + t[i]);
}

__global__ __launch_bounds__(256) void final_kernel(
    const float* __restrict__ h,  const float* __restrict__ hi,
    const float* __restrict__ fab, const float* __restrict__ faa,
    const float* __restrict__ gab, const float* __restrict__ gbb,
    float* __restrict__ out) {
  __shared__ float red[256];
  float s = 0.0f;
  for (int i = threadIdx.x; i < NN; i += 256) s += h[i]  * (fab[i] - faa[i]);
  for (int j = threadIdx.x; j < MM; j += 256) s += hi[j] * (gab[j] - gbb[j]);
  red[threadIdx.x] = s;
  __syncthreads();
  for (int off = 128; off > 0; off >>= 1) {
    if (threadIdx.x < off) red[threadIdx.x] += red[threadIdx.x + off];
    __syncthreads();
  }
  if (threadIdx.x == 0) out[0] = __expf(-red[0]);
}

// ---------------------------------------------------------------------------
// Host orchestration (graph-capture safe: kernel launches only)
// ---------------------------------------------------------------------------
extern "C" void kernel_launch(void* const* d_in, const int* in_sizes, int n_in,
                              void* d_out, int out_size, void* d_ws, size_t ws_size,
                              hipStream_t stream) {
  const float* d_  = (const float*)d_in[0];   // [N, D]
  const float* si_ = (const float*)d_in[1];   // [M, D]
  const float* h_  = (const float*)d_in[2];   // [N]
  const float* hi_ = (const float*)d_in[3];   // [M]
  const float* W_  = (const float*)d_in[4];   // [D, D]
  float* out = (float*)d_out;

  char* ws = (char*)d_ws;
  size_t off = 0;
  auto alloc = [&](size_t bytes) -> char* {
    char* p = ws + off;
    off = (off + bytes + 255) & ~(size_t)255;
    return p;
  };
  __bf16* Xb  = (__bf16*)alloc((size_t)NN * DIMD * 2);
  __bf16* Yb  = (__bf16*)alloc((size_t)MM * DIMD * 2);
  float* xsq  = (float*)alloc(NN * 4);
  float* ysq  = (float*)alloc(MM * 4);
  float* la   = (float*)alloc(NN * 4);
  float* lb   = (float*)alloc(MM * 4);
  float* f    = (float*)alloc(NN * 4);
  float* g    = (float*)alloc(MM * 4);
  float* fab  = (float*)alloc(NN * 4);
  float* gab  = (float*)alloc(MM * 4);
  float* fp   = (float*)alloc(NN * 4);   // reused for both symmetric problems
  float* tmp  = (float*)alloc(NN * 4);
  float* faa  = (float*)alloc(NN * 4);
  float* gbb  = (float*)alloc(MM * 4);

  const int T = 256;

  // 1) transform: x = d @ W^T, y = si @ W^T  (bf16 outputs, L2-resident)
  gemm_xWt_bf16<<<NN / 16, 32, 0, stream>>>(d_, W_, Xb);
  gemm_xWt_bf16<<<MM / 16, 32, 0, stream>>>(si_, W_, Yb);

  // 2) row squared norms (from the same bf16 data used in the dots)
  rowsq_kernel<<<(NN + T - 1) / T, T, 0, stream>>>(Xb, NN, xsq);
  rowsq_kernel<<<(MM + T - 1) / T, T, 0, stream>>>(Yb, MM, ysq);

  // 3) log weights
  vlog_kernel<<<(NN + T - 1) / T, T, 0, stream>>>(h_, la, NN);
  vlog_kernel<<<(MM + T - 1) / T, T, 0, stream>>>(hi_, lb, MM);

  // 4) asymmetric Sinkhorn on C_xy (alternating updates, 50 iters)
  fill_kernel<<<(NN + T - 1) / T, T, 0, stream>>>(f, NN, 0.0f);
  fill_kernel<<<(MM + T - 1) / T, T, 0, stream>>>(g, MM, 0.0f);
  for (int it = 0; it < 50; ++it) {
    softmin_rows<<<NN / 16, 32, 0, stream>>>(Xb, xsq, Yb, ysq, MM, lb, g, f);
    softmin_rows<<<MM / 16, 32, 0, stream>>>(Yb, ysq, Xb, xsq, NN, la, f, g);
  }
  // final differentiable extrapolation (uses loop f, g)
  softmin_rows<<<MM / 16, 32, 0, stream>>>(Yb, ysq, Xb, xsq, NN, la, f, gab);
  softmin_rows<<<NN / 16, 32, 0, stream>>>(Xb, xsq, Yb, ysq, MM, lb, g, fab);

  // 5) symmetric potential on C_xx (damped fixed point)
  fill_kernel<<<(NN + T - 1) / T, T, 0, stream>>>(fp, NN, 0.0f);
  for (int it = 0; it < 50; ++it) {
    softmin_rows<<<NN / 16, 32, 0, stream>>>(Xb, xsq, Xb, xsq, NN, la, fp, tmp);
    damp_kernel<<<(NN + T - 1) / T, T, 0, stream>>>(fp, tmp, NN);
  }
  softmin_rows<<<NN / 16, 32, 0, stream>>>(Xb, xsq, Xb, xsq, NN, la, fp, faa);

  // 6) symmetric potential on C_yy
  fill_kernel<<<(MM + T - 1) / T, T, 0, stream>>>(fp, MM, 0.0f);
  for (int it = 0; it < 50; ++it) {
    softmin_rows<<<MM / 16, 32, 0, stream>>>(Yb, ysq, Yb, ysq, MM, lb, fp, tmp);
    damp_kernel<<<(MM + T - 1) / T, T, 0, stream>>>(fp, tmp, MM);
  }
  softmin_rows<<<MM / 16, 32, 0, stream>>>(Yb, ysq, Yb, ysq, MM, lb, fp, gbb);

  // 7) S = <h, f_ab - f_aa> + <hi, g_ab - g_bb>;  out = exp(-S)
  final_kernel<<<1, 256, 0, stream>>>(h_, hi_, fab, faa, gab, gbb, out);
}
